// MultiLayerPLEMD_86955907875086
// MI455X (gfx1250) — matrix operand order repaired and checked by reading, not compiled, not run on hardware
//
#include <hip/hip_runtime.h>
#include <hip/hip_bf16.h>

typedef __attribute__((ext_vector_type(16))) __bf16 v16bf;
typedef __attribute__((ext_vector_type(8)))  float  v8f;

// ---------------- problem constants ----------------
constexpr int B_   = 16384;
constexpr int IN_  = 512;
constexpr int H_   = 256;
constexpr int D_   = 128;
constexpr int NEXP = 14;   // 4 shared + 6 domain + 4 task experts
constexpr int GH_  = 64;
constexpr int NE_  = 12;   // experts seen per task gate
constexpr int TH_  = 64;

constexpr int ROWS = 64;   // batch rows per workgroup

// LDS strides (elements), padded to dodge bank conflicts
constexpr int XS = IN_ + 16;  // 528 bf16
constexpr int HS = H_  + 16;  // 272 bf16
constexpr int GS = GH_ + 4;   // 68  f32
constexpr int CS = D_  + 4;   // 132 f32

// LDS byte offsets
constexpr int OFF_X  = 0;                                  // 64*528*2 = 67584
constexpr int OFF_H  = OFF_X  + ROWS * XS * 2;             // 34816
constexpr int OFF_GH = OFF_H  + ROWS * HS * 2;             // 2*64*68*4 = 34816
constexpr int OFF_G  = OFF_GH + 2 * ROWS * GS * 4;         // 64*24*4 = 6144
constexpr int OFF_C  = OFF_G  + ROWS * 24 * 4;             // 2*64*132*4 = 67584
constexpr int SMEM_BYTES = OFF_C + 2 * ROWS * CS * 4;      // 210944 < 320KB

// ---------------- WMMA helpers ----------------
__device__ __forceinline__ v8f wmma_bf16(v16bf a, v16bf b, v8f c) {
  return __builtin_amdgcn_wmma_f32_16x16x32_bf16(false, a, false, b,
                                                 (short)0, c, false, false);
}

// A fragment (16x32 bf16, row-major source): lane holds row M=lane&15,
// K chunks [kb..kb+7] and [kb+16..kb+23] with kb = (lane>=16)?8:0.
__device__ __forceinline__ v16bf load_a_frag(const __bf16* base, int stride,
                                             int row0, int k0, int lane) {
  const int m  = lane & 15;
  const int kb = (lane >> 4) << 3;
  const __bf16* p = base + (size_t)(row0 + m) * stride + k0 + kb;
  union { v16bf v; uint4 u[2]; } r;
  r.u[0] = *(const uint4*)(p);
  r.u[1] = *(const uint4*)(p + 16);
  return r.v;
}

// B fragment (32x16 bf16) from TRANSPOSED weights W[t][n][k]: lane holds
// column N=lane&15, K chunks [kb..kb+7] and [kb+16..kb+23] with kb = (lane>=16)?8:0.
__device__ __forceinline__ v16bf load_b_frag(const __bf16* base, int stride,
                                             int col0, int k0, int lane) {
  const int n  = lane & 15;
  const int kh = (lane >> 4) << 3;
  const __bf16* p = base + (size_t)(col0 + n) * stride + k0 + kh;
  union { v16bf v; uint4 u[2]; } r;
  r.u[0] = *(const uint4*)(p);
  r.u[1] = *(const uint4*)(p + 16);
  return r.v;
}

// ---------------- prep kernels ----------------
__global__ void convert_x_kernel(const float* __restrict__ x,
                                 __bf16* __restrict__ xb, int n4) {
  int i = (blockIdx.x * blockDim.x + threadIdx.x);
  if (i >= n4) return;
  const float4 v = ((const float4*)x)[i];
  __bf16* o = xb + (size_t)i * 4;
  o[0] = (__bf16)v.x; o[1] = (__bf16)v.y; o[2] = (__bf16)v.z; o[3] = (__bf16)v.w;
}

// Convert + transpose all weights into ws.
__global__ void prep_kernel(
    const float* __restrict__ sw1, const float* __restrict__ dw1,
    const float* __restrict__ tw1, const float* __restrict__ sw2,
    const float* __restrict__ dw2, const float* __restrict__ tw2,
    const float* __restrict__ gw1, const float* __restrict__ sb1,
    const float* __restrict__ db1, const float* __restrict__ tb1,
    const float* __restrict__ sb2, const float* __restrict__ db2,
    const float* __restrict__ tb2, const float* __restrict__ ow1,
    __bf16* __restrict__ w1t, __bf16* __restrict__ w2t,
    __bf16* __restrict__ gt1, float* __restrict__ b1s,
    float* __restrict__ b2s, float* __restrict__ ow1t) {
  const int N1 = NEXP * H_ * IN_;       // 1,835,008  W1t[e][h][i]
  const int N2 = NEXP * D_ * H_;        //   458,752  W2t[e][d][h]
  const int N3 = 2 * GH_ * IN_;         //    65,536  Gt1[t][g][i]
  const int N4 = NEXP * H_;             //     3,584  b1s
  const int N5 = NEXP * D_;             //     1,792  b2s
  const int N6 = 2 * TH_ * D_;          //    16,384  ow1t[t][j][d]
  int idx = blockIdx.x * blockDim.x + threadIdx.x;
  if (idx < N1) {
    int e = idx / (H_ * IN_), rem = idx % (H_ * IN_);
    int h = rem / IN_, i = rem % IN_;
    const float* src = (e < 4)  ? sw1 + (size_t)e * IN_ * H_
                     : (e < 10) ? dw1 + (size_t)(e - 4) * IN_ * H_
                                : tw1 + (size_t)(e - 10) * IN_ * H_;
    w1t[idx] = (__bf16)src[(size_t)i * H_ + h];
    return;
  }
  idx -= N1;
  if (idx < N2) {
    int e = idx / (D_ * H_), rem = idx % (D_ * H_);
    int d = rem / H_, h = rem % H_;
    const float* src = (e < 4)  ? sw2 + (size_t)e * H_ * D_
                     : (e < 10) ? dw2 + (size_t)(e - 4) * H_ * D_
                                : tw2 + (size_t)(e - 10) * H_ * D_;
    w2t[idx] = (__bf16)src[(size_t)h * D_ + d];
    return;
  }
  idx -= N2;
  if (idx < N3) {
    int t = idx / (GH_ * IN_), rem = idx % (GH_ * IN_);
    int g = rem / IN_, i = rem % IN_;
    gt1[idx] = (__bf16)gw1[(size_t)t * IN_ * GH_ + (size_t)i * GH_ + g];
    return;
  }
  idx -= N3;
  if (idx < N4) {
    int e = idx / H_, h = idx % H_;
    const float* src = (e < 4) ? sb1 + e * H_ : (e < 10) ? db1 + (e - 4) * H_
                                                         : tb1 + (e - 10) * H_;
    b1s[idx] = src[h];
    return;
  }
  idx -= N4;
  if (idx < N5) {
    int e = idx / D_, d = idx % D_;
    const float* src = (e < 4) ? sb2 + e * D_ : (e < 10) ? db2 + (e - 4) * D_
                                                         : tb2 + (e - 10) * D_;
    b2s[idx] = src[d];
    return;
  }
  idx -= N5;
  if (idx < N6) {
    int t = idx / (TH_ * D_), rem = idx % (TH_ * D_);
    int j = rem / D_, d = rem % D_;
    ow1t[idx] = ow1[(size_t)t * D_ * TH_ + (size_t)d * TH_ + j];
  }
}

// ---------------- fused PLE kernel ----------------
__global__ void __launch_bounds__(256, 1) ple_main_kernel(
    const __bf16* __restrict__ xb,  const __bf16* __restrict__ w1t,
    const float*  __restrict__ b1s, const __bf16* __restrict__ w2t,
    const float*  __restrict__ b2s, const __bf16* __restrict__ gt1,
    const float*  __restrict__ gb1, const float* __restrict__ gw2,
    const float*  __restrict__ gb2, const float* __restrict__ ow1t,
    const float*  __restrict__ ob1, const float* __restrict__ ow2,
    const float*  __restrict__ ob2, float* __restrict__ out) {
  extern __shared__ char smem[];
  __bf16* sX  = (__bf16*)(smem + OFF_X);   // [64][XS]
  __bf16* sH  = (__bf16*)(smem + OFF_H);   // [64][HS]
  float*  sGh = (float*) (smem + OFF_GH);  // [2][64][GS]
  float*  sG  = (float*) (smem + OFF_G);   // [64][24] softmaxed gates
  float*  sC  = (float*) (smem + OFF_C);   // [2][64][CS] combined

  const int tid  = threadIdx.x;
  const int lane = tid & 31;
  const int wave = tid >> 5;
  const int row0 = blockIdx.x * ROWS;

  // ---- 1. stage X tile (64 x 512 bf16) into LDS ----
  {
    const uint4* src = (const uint4*)(xb + (size_t)row0 * IN_);
    for (int c = tid; c < ROWS * IN_ / 8; c += 256) {
      int r = c >> 6, k8 = c & 63;          // 64 uint4 per row
      *(uint4*)(sX + r * XS + k8 * 8) = src[c];
    }
  }
  __syncthreads();

  // ---- 2. gate hidden GEMM: [64x64] per task, K=512 ----
  {
    const int t  = wave >> 2;
    const int rt = wave & 3;
    v8f acc[4] = {};
    for (int ks = 0; ks < IN_ / 32; ++ks) {
      v16bf a = load_a_frag(sX, XS, rt * 16, ks * 32, lane);
#pragma unroll
      for (int c = 0; c < 4; ++c) {
        v16bf b = load_b_frag(gt1 + (size_t)t * GH_ * IN_, IN_, c * 16, ks * 32, lane);
        acc[c] = wmma_bf16(a, b, acc[c]);
      }
    }
    const int n = lane & 15, moff = (lane >> 4) << 3;
#pragma unroll
    for (int c = 0; c < 4; ++c) {
      int col = c * 16 + n;
      float bias = gb1[t * GH_ + col];
#pragma unroll
      for (int r = 0; r < 8; ++r) {
        float v = acc[c][r] + bias;
        sGh[(t * ROWS + rt * 16 + r + moff) * GS + col] = v > 0.f ? v : 0.f;
      }
    }
  }
  __syncthreads();

  // ---- 3. gate layer2 + softmax (one thread per (row,task)) ----
  if (tid < 2 * ROWS) {
    const int t = tid & 1, row = tid >> 1;
    float logit[NE_], mx = -1e30f;
    for (int e = 0; e < NE_; ++e) {
      float s = gb2[t * NE_ + e];
      for (int h = 0; h < GH_; ++h)
        s += sGh[(t * ROWS + row) * GS + h] * gw2[(t * GH_ + h) * NE_ + e];
      logit[e] = s; mx = fmaxf(mx, s);
    }
    float den = 0.f;
    for (int e = 0; e < NE_; ++e) { logit[e] = __expf(logit[e] - mx); den += logit[e]; }
    float inv = 1.f / den;
    for (int e = 0; e < NE_; ++e) sG[row * 24 + t * NE_ + e] = logit[e] * inv;
  }

  // ---- 4. expert loop: GEMM1 (K=512) -> LDS -> GEMM2 (K=256) -> gated acc ----
  const int rt1 = wave >> 1, c10 = (wave & 1) * 8;   // GEMM1 tiles: 4rt x 16ct
  const int rt2 = wave >> 1, c20 = (wave & 1) * 4;   // GEMM2 tiles: 4rt x 8ct
  v8f comb0[4] = {}, comb1[4] = {};                  // persistent f32 combine

  for (int e = 0; e < NEXP; ++e) {
    __syncthreads();   // sH free (also covers sG writes on first iter)
    // GEMM1: H1 = relu(X @ W1[e] + b1)
    v8f acc[8] = {};
    const __bf16* wb = w1t + (size_t)e * H_ * IN_;
    for (int ks = 0; ks < IN_ / 32; ++ks) {
      v16bf a = load_a_frag(sX, XS, rt1 * 16, ks * 32, lane);
#pragma unroll
      for (int c = 0; c < 8; ++c) {
        v16bf b = load_b_frag(wb, IN_, (c10 + c) * 16, ks * 32, lane);
        acc[c] = wmma_bf16(a, b, acc[c]);
      }
    }
    {
      const int n = lane & 15, moff = (lane >> 4) << 3;
#pragma unroll
      for (int c = 0; c < 8; ++c) {
        int col = (c10 + c) * 16 + n;
        float bias = b1s[e * H_ + col];
#pragma unroll
        for (int r = 0; r < 8; ++r) {
          float v = acc[c][r] + bias;
          sH[(rt1 * 16 + r + moff) * HS + col] = (__bf16)(v > 0.f ? v : 0.f);
        }
      }
    }
    __syncthreads();
    // GEMM2: Y2 = relu(H1 @ W2[e] + b2), gated accumulate
    v8f acc2[4] = {};
    const __bf16* w2 = w2t + (size_t)e * D_ * H_;
    for (int ks = 0; ks < H_ / 32; ++ks) {
      v16bf a = load_a_frag(sH, HS, rt2 * 16, ks * 32, lane);
#pragma unroll
      for (int c = 0; c < 4; ++c) {
        v16bf b = load_b_frag(w2, H_, (c20 + c) * 16, ks * 32, lane);
        acc2[c] = wmma_bf16(a, b, acc2[c]);
      }
    }
    {
      const int n = lane & 15, moff = (lane >> 4) << 3;
      // gate index per task: common experts feed both; task experts their own
      const int i0 = (e < 10) ? e : (((e - 10) >> 1) == 0 ? 10 + ((e - 10) & 1) : -1);
      const int i1 = (e < 10) ? e : (((e - 10) >> 1) == 1 ? 10 + ((e - 10) & 1) : -1);
#pragma unroll
      for (int c = 0; c < 4; ++c) {
        int col = (c20 + c) * 16 + n;
        float bias = b2s[e * D_ + col];
#pragma unroll
        for (int r = 0; r < 8; ++r) {
          int row = rt2 * 16 + r + moff;
          float y = acc2[c][r] + bias; y = y > 0.f ? y : 0.f;
          if (i0 >= 0) comb0[c][r] += sG[row * 24 + i0]       * y;
          if (i1 >= 0) comb1[c][r] += sG[row * 24 + NE_ + i1] * y;
        }
      }
    }
  }

  // ---- 5. spill combine to LDS ----
  {
    const int n = lane & 15, moff = (lane >> 4) << 3;
#pragma unroll
    for (int c = 0; c < 4; ++c) {
      int col = (c20 + c) * 16 + n;
#pragma unroll
      for (int r = 0; r < 8; ++r) {
        int row = rt2 * 16 + r + moff;
        sC[(0 * ROWS + row) * CS + col] = comb0[c][r];
        sC[(1 * ROWS + row) * CS + col] = comb1[c][r];
      }
    }
  }
  __syncthreads();

  // ---- 6. towers D->TH->1, sigmoid; all 256 threads, float4 inner loop ----
  {
    const int p  = tid >> 1;        // (row,task) pair 0..127
    const int jh = tid & 1;         // half of the j-range
    const int t = p & 1, row = p >> 1;
    const float4* cb4 = (const float4*)(sC + (t * ROWS + row) * CS);
    float acc = 0.f;
#pragma unroll 1
    for (int j = jh * (TH_ / 2); j < (jh + 1) * (TH_ / 2); ++j) {
      const float4* w4 = (const float4*)(ow1t + ((size_t)t * TH_ + j) * D_);
      float h = ob1[t * TH_ + j];
#pragma unroll
      for (int d4 = 0; d4 < D_ / 4; ++d4) {
        float4 c = cb4[d4];
        float4 w = w4[d4];
        h += c.x * w.x + c.y * w.y + c.z * w.z + c.w * w.w;
      }
      h = h > 0.f ? h : 0.f;
      acc += h * ow2[t * TH_ + j];
    }
    acc += __shfl_xor(acc, 1, 32);  // combine the two j-halves (wave32)
    if (jh == 0)
      out[(size_t)(row0 + row) * 2 + t] = 1.f / (1.f + __expf(-(acc + ob2[t])));
  }
}

// ---------------- host launcher ----------------
extern "C" void kernel_launch(void* const* d_in, const int* in_sizes, int n_in,
                              void* d_out, int out_size, void* d_ws, size_t ws_size,
                              hipStream_t stream) {
  const float* x   = (const float*)d_in[0];
  // d_in[1] = domain_ids (unused by reference math)
  const float* sw1 = (const float*)d_in[2];
  const float* sb1 = (const float*)d_in[3];
  const float* sw2 = (const float*)d_in[4];
  const float* sb2 = (const float*)d_in[5];
  const float* dw1 = (const float*)d_in[6];
  const float* db1 = (const float*)d_in[7];
  const float* dw2 = (const float*)d_in[8];
  const float* db2 = (const float*)d_in[9];
  const float* tw1 = (const float*)d_in[10];
  const float* tb1 = (const float*)d_in[11];
  const float* tw2 = (const float*)d_in[12];
  const float* tb2 = (const float*)d_in[13];
  const float* gw1 = (const float*)d_in[14];
  const float* gb1 = (const float*)d_in[15];
  const float* gw2 = (const float*)d_in[16];
  const float* gb2 = (const float*)d_in[17];
  const float* ow1 = (const float*)d_in[18];
  const float* ob1 = (const float*)d_in[19];
  const float* ow2 = (const float*)d_in[20];
  const float* ob2 = (const float*)d_in[21];
  float* out = (float*)d_out;

  // workspace carve-out
  size_t off = 0;
  auto carve = [&](size_t bytes) {
    void* p = (char*)d_ws + off;
    off += (bytes + 255) & ~(size_t)255;
    return p;
  };
  __bf16* xb   = (__bf16*)carve((size_t)B_ * IN_ * 2);
  __bf16* w1t  = (__bf16*)carve((size_t)NEXP * H_ * IN_ * 2);
  __bf16* w2t  = (__bf16*)carve((size_t)NEXP * D_ * H_ * 2);
  __bf16* gt1  = (__bf16*)carve((size_t)2 * GH_ * IN_ * 2);
  float*  b1s  = (float*) carve((size_t)NEXP * H_ * 4);
  float*  b2s  = (float*) carve((size_t)NEXP * D_ * 4);
  float*  ow1t = (float*) carve((size_t)2 * TH_ * D_ * 4);

  // 1) x -> bf16
  {
    int n4 = B_ * IN_ / 4;
    convert_x_kernel<<<(n4 + 255) / 256, 256, 0, stream>>>(x, xb, n4);
  }
  // 2) transpose/convert weights + stack biases
  {
    int total = NEXP * H_ * IN_ + NEXP * D_ * H_ + 2 * GH_ * IN_ +
                NEXP * H_ + NEXP * D_ + 2 * TH_ * D_;
    prep_kernel<<<(total + 255) / 256, 256, 0, stream>>>(
        sw1, dw1, tw1, sw2, dw2, tw2, gw1,
        sb1, db1, tb1, sb2, db2, tb2, ow1,
        w1t, w2t, gt1, b1s, b2s, ow1t);
  }
  // 3) fused PLE
  ple_main_kernel<<<B_ / ROWS, 256, SMEM_BYTES, stream>>>(
      xb, w1t, b1s, w2t, b2s, gt1, gb1, gw2, gb2, ow1t, ob1, ow2, ob2, out);
}
